// Attention_CA_86320252715168
// MI455X (gfx1250) — matrix operand order repaired and checked by reading, not compiled, not run on hardware
//
#include <hip/hip_runtime.h>
#include <hip/hip_bf16.h>

// ---------------------------------------------------------------------------
// Cross-attention block for MI455X (gfx1250), wave32 WMMA f16->f32 pipeline.
// Attention stage: K/V tiles staged to LDS via gfx1250 async-to-LDS copies,
// shared by all 4 waves in the block (4x cut in L2 K/V traffic).
// ---------------------------------------------------------------------------

typedef _Float16 h16v __attribute__((ext_vector_type(16)));
typedef _Float16 h8v  __attribute__((ext_vector_type(8)));
typedef float    f8v  __attribute__((ext_vector_type(8)));
typedef int      i4v  __attribute__((ext_vector_type(4)));

#define DIM      768
#define NHEADS   12
#define HDIM     64
#define BATCH    4
#define NQ       1024
#define MK       2048
#define QK_SCALE 0.125f   /* 64^-0.5 */
#define MASK_NEG -9.0e15f

// ---- gfx1250 async global->LDS copy (ASYNCcnt path), with fallback ---------
#if defined(__has_builtin)
#  if __has_builtin(__builtin_amdgcn_global_load_async_to_lds_b128)
#    define HAVE_ASYNC_LDS 1
#  endif
#endif
#ifndef HAVE_ASYNC_LDS
#  define HAVE_ASYNC_LDS 0
#endif

#if HAVE_ASYNC_LDS
typedef __attribute__((address_space(1))) i4v* g_i4p;   // global (device) ptr
typedef __attribute__((address_space(3))) i4v* l_i4p;   // LDS ptr
#endif

__device__ __forceinline__ void stage16(const _Float16* g, _Float16* l) {
#if HAVE_ASYNC_LDS
  __builtin_amdgcn_global_load_async_to_lds_b128((g_i4p)g, (l_i4p)l, 0, 0);
#else
  *(uint4*)l = *(const uint4*)g;
#endif
}

__device__ __forceinline__ void stage_drain() {
#if HAVE_ASYNC_LDS
  asm volatile("s_wait_asynccnt 0" ::: "memory");
#endif
}

// ---- WMMA fragment helpers (A and B share contiguous k order, wave32) ----------

__device__ __forceinline__ f8v wmma_f16(h16v a, h16v b, f8v c) {
  // D = A(16x32 f16) * B(32x16 f16) + C(16x16 f32)
  return __builtin_amdgcn_wmma_f32_16x16x32_f16(
      /*neg_a=*/false, a, /*neg_b=*/false, b,
      /*c_mod=*/(short)0, c, /*reuse_a=*/false, /*reuse_b=*/false);
}

// A fragment: 16x32 tile, row-major source with leading dim lda (elements).
// Lane L holds row L%16; contiguous k-chunk [koff, koff+16),
// koff = (L/16)*16.  Two 16B loads per lane.
__device__ __forceinline__ h16v load_a_frag(const _Float16* A, int lda) {
  int lane = threadIdx.x & 31;
  const _Float16* p = A + (lane & 15) * lda + ((lane >> 4) << 4);
  union { h16v v; h8v h[2]; } u;
  u.h[0] = *(const h8v*)(p);
  u.h[1] = *(const h8v*)(p + 8);
  return u.v;
}

// B fragment: 32x16 tile where B[k][n] = Bm[n][k] (Bm row-major, ld = ldb).
// Lane L holds column n = L%16, contiguous k-chunk of 16 at (L/16)*16.
// One 32B load per lane.
__device__ __forceinline__ h16v load_b_frag(const _Float16* Bm, int ldb) {
  int lane = threadIdx.x & 31;
  return *(const h16v*)(Bm + (lane & 15) * ldb + ((lane >> 4) << 4));
}

// ---- Kernel 0: f32 -> f16 conversion ---------------------------------------

__global__ void cvt_f32_f16(const float* __restrict__ in,
                            _Float16* __restrict__ out, int n) {
  int i = blockIdx.x * blockDim.x + threadIdx.x;
  if (i < n) out[i] = (_Float16)in[i];
}

// ---- Kernel 1: KV projection -----------------------------------------------
// kvp = kv @ Wkv^T  (8192 x 1536, K=768)
// c2 <  768 -> K head tensor  Kf[b,h,m,d]   (row-major in d)
// c2 >= 768 -> V head tensor  Vt[b,h,d,m]   (transposed: contiguous in m)
__global__ void kvproj_kernel(const _Float16* __restrict__ A,
                              const _Float16* __restrict__ W,
                              _Float16* __restrict__ Kf,
                              _Float16* __restrict__ Vt) {
  const int Kd = DIM;
  int wave = threadIdx.x >> 5, lane = threadIdx.x & 31;
  int mBase = blockIdx.x * 128 + wave * 16;
  int nBase = blockIdx.y * 64;
  f8v acc[4] = {};
  for (int k = 0; k < Kd; k += 32) {
    h16v af = load_a_frag(A + (size_t)mBase * Kd + k, Kd);
    __builtin_prefetch(A + (size_t)mBase * Kd + k + 32, 0, 1);
#pragma unroll
    for (int t = 0; t < 4; ++t) {
      h16v bf = load_b_frag(W + (size_t)(nBase + t * 16) * Kd + k, Kd);
      acc[t] = wmma_f16(af, bf, acc[t]);
    }
  }
  int rowg = (lane >> 4) << 3, col = lane & 15;
#pragma unroll
  for (int t = 0; t < 4; ++t) {
    int c2 = nBase + t * 16 + col;
#pragma unroll
    for (int r = 0; r < 8; ++r) {
      int gm = mBase + rowg + r;         // 0 .. 8191
      int b = gm / MK, mm = gm - b * MK;
      _Float16 val = (_Float16)acc[t][r];
      if (c2 < DIM) {
        int h = c2 / HDIM, dd = c2 - h * HDIM;
        Kf[(((b * NHEADS + h) * MK) + mm) * HDIM + dd] = val;
      } else {
        int c = c2 - DIM;
        int h = c / HDIM, dd = c - h * HDIM;
        Vt[(((b * NHEADS + h) * HDIM) + dd) * MK + mm] = val;
      }
    }
  }
}

// ---- Kernel 2: Q projection ------------------------------------------------
// qp = (q @ Wq^T) * SCALE  (4096 x 768) -> Qf[b,h,n,d]
__global__ void qproj_kernel(const _Float16* __restrict__ A,
                             const _Float16* __restrict__ W,
                             _Float16* __restrict__ Qf) {
  const int Kd = DIM;
  int wave = threadIdx.x >> 5, lane = threadIdx.x & 31;
  int mBase = blockIdx.x * 128 + wave * 16;
  int nBase = blockIdx.y * 64;
  f8v acc[4] = {};
  for (int k = 0; k < Kd; k += 32) {
    h16v af = load_a_frag(A + (size_t)mBase * Kd + k, Kd);
#pragma unroll
    for (int t = 0; t < 4; ++t) {
      h16v bf = load_b_frag(W + (size_t)(nBase + t * 16) * Kd + k, Kd);
      acc[t] = wmma_f16(af, bf, acc[t]);
    }
  }
  int rowg = (lane >> 4) << 3, col = lane & 15;
#pragma unroll
  for (int t = 0; t < 4; ++t) {
    int c = nBase + t * 16 + col;
    int h = c / HDIM, dd = c - h * HDIM;
#pragma unroll
    for (int r = 0; r < 8; ++r) {
      int gm = mBase + rowg + r;         // 0 .. 4095
      int b = gm / NQ, nn = gm - b * NQ;
      Qf[(((b * NHEADS + h) * NQ) + nn) * HDIM + dd] =
          (_Float16)(acc[t][r] * QK_SCALE);
    }
  }
}

// ---- Kernel 3: flash attention --------------------------------------------
// 4 waves per block, all on the same (b,h); each wave owns one 16-row q tile.
// Per 32-key chunk:
//   - block cooperatively async-copies K tile (32x64) + Vt tile (64x32) to LDS
//   - each wave: 4 WMMA for S, mask, online softmax, P repack via LDS,
//     4 WMMA for PV into O(16x64).
#define ATT_WAVES 4
__global__ void attn_kernel(const _Float16* __restrict__ Qf,
                            const _Float16* __restrict__ Kf,
                            const _Float16* __restrict__ Vt,
                            const int* __restrict__ mask,
                            _Float16* __restrict__ Xf) {
  __shared__ _Float16 kT[32 * HDIM];              // K chunk, row-major in d
  __shared__ _Float16 vT[HDIM * 32];              // Vt chunk, row-major in keys
  __shared__ _Float16 pT[ATT_WAVES][16 * 32];     // per-wave P repack tile
  int tid = threadIdx.x;
  int wave = tid >> 5, lane = tid & 31;
  int bh = blockIdx.y;                       // b*H + h
  int b = bh / NHEADS, h = bh - b * NHEADS;
  int nBase = (blockIdx.x * ATT_WAVES + wave) * 16;

  const _Float16* Qp = Qf + ((size_t)bh * NQ + nBase) * HDIM;
  const _Float16* Kp = Kf + (size_t)bh * MK * HDIM;
  const _Float16* Vp = Vt + (size_t)bh * HDIM * MK;
  const int* mp = mask + ((size_t)b * NQ + nBase) * MK;

  // Q A-fragments (16 x 64 = two 32-wide k chunks), loaded once.
  h16v qa0 = load_a_frag(Qp, HDIM);
  h16v qa1 = load_a_frag(Qp + 32, HDIM);

  f8v o[4] = {};
  float mrow[8], lrow[8];
#pragma unroll
  for (int r = 0; r < 8; ++r) { mrow[r] = -3.0e38f; lrow[r] = 0.0f; }

  int rowg = (lane >> 4) << 3, col = lane & 15;
  _Float16* Ptile = pT[wave];

  for (int kb = 0; kb < MK; kb += 32) {
    // ---- stage K (contiguous 4KB) and V (64 rows x 64B) into LDS ----
    // 256 16B chunks each; 128 threads -> 2 chunks per tensor per thread.
#pragma unroll
    for (int c = tid; c < 256; c += 128) {
      stage16(Kp + (size_t)kb * HDIM + c * 8, kT + c * 8);
      stage16(Vp + (size_t)(c >> 2) * MK + kb + (c & 3) * 8, vT + c * 8);
    }
    stage_drain();
    __syncthreads();

    // ---- scores S = Q * K^T for 32 keys (two 16x16 C tiles) ----
    f8v s0 = {}, s1 = {};
    s0 = wmma_f16(qa0, load_b_frag(kT, HDIM), s0);
    s0 = wmma_f16(qa1, load_b_frag(kT + 32, HDIM), s0);
    s1 = wmma_f16(qa0, load_b_frag(kT + 16 * HDIM, HDIM), s1);
    s1 = wmma_f16(qa1, load_b_frag(kT + 16 * HDIM + 32, HDIM), s1);

    // ---- mask + online softmax (rows live in 16-lane halves) ----
#pragma unroll
    for (int r = 0; r < 8; ++r) {
      const int* mr = mp + (size_t)(rowg + r) * MK + kb;
      s0[r] = (mr[col] == 0) ? MASK_NEG : s0[r];
      s1[r] = (mr[col + 16] == 0) ? MASK_NEG : s1[r];
    }
#pragma unroll
    for (int r = 0; r < 8; ++r) {
      float mx = fmaxf(s0[r], s1[r]);
#pragma unroll
      for (int sh = 8; sh; sh >>= 1) mx = fmaxf(mx, __shfl_xor(mx, sh, 16));
      float mnew = fmaxf(mrow[r], mx);
      float p0 = __expf(s0[r] - mnew);
      float p1 = __expf(s1[r] - mnew);
      s0[r] = p0; s1[r] = p1;
      float rs = p0 + p1;
#pragma unroll
      for (int sh = 8; sh; sh >>= 1) rs += __shfl_xor(rs, sh, 16);
      float corr = __expf(mrow[r] - mnew);
      lrow[r] = lrow[r] * corr + rs;
      mrow[r] = mnew;
#pragma unroll
      for (int t = 0; t < 4; ++t) o[t][r] *= corr;
    }

    // ---- C-layout P -> LDS (16x32 row-major f16) -> A-fragment ----
#pragma unroll
    for (int r = 0; r < 8; ++r) {
      Ptile[(rowg + r) * 32 + col]      = (_Float16)s0[r];
      Ptile[(rowg + r) * 32 + 16 + col] = (_Float16)s1[r];
    }
    asm volatile("s_wait_dscnt 0" ::: "memory");   // DS stores visible in-wave
    h16v pa = load_a_frag(Ptile, 32);

    // ---- O += P * V  (vT contiguous along keys -> clean B frags) ----
#pragma unroll
    for (int t = 0; t < 4; ++t) {
      h16v vb = load_b_frag(vT + (size_t)(t * 16) * 32, 32);
      o[t] = wmma_f16(pa, vb, o[t]);
    }
    __syncthreads();   // protect kT/vT before next chunk's staging
  }

  // ---- normalize and write X[b, n, h*64 + d] ----
#pragma unroll
  for (int r = 0; r < 8; ++r) {
    float inv = 1.0f / lrow[r];
    int n = nBase + rowg + r;
#pragma unroll
    for (int t = 0; t < 4; ++t) {
      Xf[((size_t)b * NQ + n) * DIM + h * HDIM + t * 16 + col] =
          (_Float16)(o[t][r] * inv);
    }
  }
}

// ---- Kernel 4: output projection + bias (f32 out) --------------------------
__global__ void outproj_kernel(const _Float16* __restrict__ A,
                               const _Float16* __restrict__ W,
                               const float* __restrict__ bias,
                               float* __restrict__ out) {
  const int Kd = DIM;
  int wave = threadIdx.x >> 5, lane = threadIdx.x & 31;
  int mBase = blockIdx.x * 128 + wave * 16;
  int nBase = blockIdx.y * 64;
  f8v acc[4] = {};
  for (int k = 0; k < Kd; k += 32) {
    h16v af = load_a_frag(A + (size_t)mBase * Kd + k, Kd);
#pragma unroll
    for (int t = 0; t < 4; ++t) {
      h16v bf = load_b_frag(W + (size_t)(nBase + t * 16) * Kd + k, Kd);
      acc[t] = wmma_f16(af, bf, acc[t]);
    }
  }
  int rowg = (lane >> 4) << 3, col = lane & 15;
#pragma unroll
  for (int t = 0; t < 4; ++t) {
    int n = nBase + t * 16 + col;
    float bv = bias[n];
#pragma unroll
    for (int r = 0; r < 8; ++r) {
      out[(size_t)(mBase + rowg + r) * DIM + n] = acc[t][r] + bv;
    }
  }
}

// ---------------------------------------------------------------------------

extern "C" void kernel_launch(void* const* d_in, const int* in_sizes, int n_in,
                              void* d_out, int out_size, void* d_ws, size_t ws_size,
                              hipStream_t stream) {
  const float* q     = (const float*)d_in[0];   // [4,1024,768]
  const float* kv    = (const float*)d_in[1];   // [4,2048,768]
  const float* Wq    = (const float*)d_in[2];   // [768,768]
  const float* Wkv   = (const float*)d_in[3];   // [1536,768]
  const float* Wproj = (const float*)d_in[4];   // [768,768]
  const float* bproj = (const float*)d_in[5];   // [768]
  const int*   mask  = (const int*)d_in[6];     // [4,1,1024,2048]
  float* out = (float*)d_out;                   // [4,1024,768]

  // workspace layout (bytes, all 256-aligned)
  const size_t KV16   = 0;                         // 8192*768*2  = 12582912
  const size_t Q16    = 12582912;                  // 4096*768*2  =  6291456
  const size_t WQ16   = 18874368;                  // 768*768*2   =  1179648
  const size_t WKV16  = 20054016;                  // 1536*768*2  =  2359296
  const size_t WP16   = 22413312;                  // 768*768*2   =  1179648
  const size_t KF     = 23592960;                  // 4*12*2048*64*2 = 12582912
  const size_t VT     = 36175872;                  // 12582912
  const size_t QP     = 48758784;                  // 4*12*1024*64*2 = 6291456
  const size_t XF     = 55050240;                  // 4096*768*2  =  6291456
  const size_t TOTAL  = 61341696;
  if (ws_size < TOTAL) return;

  char* ws = (char*)d_ws;
  _Float16* kv16  = (_Float16*)(ws + KV16);
  _Float16* q16   = (_Float16*)(ws + Q16);
  _Float16* wq16  = (_Float16*)(ws + WQ16);
  _Float16* wkv16 = (_Float16*)(ws + WKV16);
  _Float16* wp16  = (_Float16*)(ws + WP16);
  _Float16* kf    = (_Float16*)(ws + KF);
  _Float16* vt    = (_Float16*)(ws + VT);
  _Float16* qp    = (_Float16*)(ws + QP);
  _Float16* xf    = (_Float16*)(ws + XF);

  // 0) f32 -> f16 conversions
  auto cvt = [&](const float* src, _Float16* dst, int n) {
    cvt_f32_f16<<<(n + 255) / 256, 256, 0, stream>>>(src, dst, n);
  };
  cvt(kv,    kv16,  BATCH * MK * DIM);
  cvt(q,     q16,   BATCH * NQ * DIM);
  cvt(Wq,    wq16,  DIM * DIM);
  cvt(Wkv,   wkv16, 2 * DIM * DIM);
  cvt(Wproj, wp16,  DIM * DIM);

  // 1) KV projection: (8192 x 1536, K=768)
  kvproj_kernel<<<dim3(BATCH * MK / 128, 2 * DIM / 64), 256, 0, stream>>>(
      kv16, wkv16, kf, vt);

  // 2) Q projection: (4096 x 768, K=768), pre-scaled
  qproj_kernel<<<dim3(BATCH * NQ / 128, DIM / 64), 256, 0, stream>>>(
      q16, wq16, qp);

  // 3) flash attention: grid = (q-tiles / ATT_WAVES, B*H)
  attn_kernel<<<dim3(NQ / 16 / ATT_WAVES, BATCH * NHEADS),
                ATT_WAVES * 32, 0, stream>>>(qp, kf, vt, mask, xf);

  // 4) output projection + bias -> f32 out
  outproj_kernel<<<dim3(BATCH * NQ / 128, DIM / 64), 256, 0, stream>>>(
      xf, wp16, bproj, out);
}
